// net_58076547776834
// MI455X (gfx1250) — matrix operand order repaired and checked by reading, not compiled, not run on hardware
//
#include <hip/hip_runtime.h>
#include <hip/hip_bf16.h>

// ---------------------------------------------------------------------------
// MI455X (gfx1250) implementation.
//  - bf16 Ksub (fits 192MB L2), v_wmma_f32_16x16x32_bf16 for the grad GEMMs
//  - v_wmma_f32_16x16x4_f32 for the fp32 MLP encoder
//  - wave32 shuffles for the row-norm reduction
//  - deterministic K-split (separate partial buffers, fixed-order reduce)
// ---------------------------------------------------------------------------

typedef __attribute__((ext_vector_type(16))) __bf16          v16bf;
typedef __attribute__((ext_vector_type(8)))  float           v8f;
typedef __attribute__((ext_vector_type(2)))  float           v2f;
typedef __attribute__((ext_vector_type(8)))  unsigned short  u16x8;

#define DEVINL __device__ __forceinline__

constexpr int N_  = 8192;   // nodes
constexpr int D_  = 32;     // feature dim
constexpr int KN_ = 3;      // kernels
constexpr int B_  = 4096;   // sampled nodes
constexpr int S1_ = 2048;   // target nodes
constexpr int H_  = 512;    // hidden
constexpr int E_  = 64;     // embedding
constexpr int D1  = D_ * KN_;        // 96
constexpr int D2  = D1 * KN_;        // 288
constexpr int DOUT = D_ + D1 + D2;   // 416
constexpr int KSPLIT = 4;            // K-dim split of grad GEMMs (deterministic)

union FragBf { v16bf v; u16x8 h[2]; };

DEVINL unsigned short f2bf(float f) {        // round-to-nearest-even f32->bf16
  unsigned u = __float_as_uint(f);
  u += 0x7FFFu + ((u >> 16) & 1u);
  return (unsigned short)(u >> 16);
}

// ---------------------------------------------------------------------------
// 1) diffusion restricted to sampled rows:  xg[j,:] = x[r,:] - L[r,:] @ x,
//    r = n_id[j].  One wave per j: lane = feature d (coalesced x reads,
//    broadcast L read).  Writes transposed bf16 xT[d][j] for the WMMA
//    B-operand, and the vec-normalized order-0 block of `out` for j < S1.
// ---------------------------------------------------------------------------
__global__ void k_diffuse(const float* __restrict__ x, const float* __restrict__ L,
                          const int* __restrict__ n_id,
                          unsigned short* __restrict__ xT, float* __restrict__ out) {
  int wave = (blockIdx.x * blockDim.x + threadIdx.x) >> 5;
  int lane = threadIdx.x & 31;
  if (wave >= B_) return;
  int r = n_id[wave];
  const float* Lr = L + (long)r * N_;
  float acc = 0.f;
#pragma unroll 4
  for (int t = 0; t < N_; ++t)
    acc = fmaf(Lr[t], x[t * D_ + lane], acc);
  float xv = x[(long)r * D_ + lane] - acc;          // TAU = 1
  xT[lane * B_ + wave] = f2bf(xv);
  if (wave < S1_) {
    float ss = xv * xv;
#pragma unroll
    for (int m = 16; m >= 1; m >>= 1) ss += __shfl_xor(ss, m, 32);
    out[wave * DOUT + lane] = xv / (sqrtf(ss) + 1e-12f);
  }
}

// ---------------------------------------------------------------------------
// 2) Ksub[k,i,j] = bf16(kernels[k, n_id[i], n_id[j]]).  100.7MB, L2-resident.
// ---------------------------------------------------------------------------
__global__ void k_gather(const float* __restrict__ kernels, const int* __restrict__ n_id,
                         unsigned short* __restrict__ Ksub) {
  int i = blockIdx.x, kk = blockIdx.y;
  int r = n_id[i];
  const float* src = kernels + ((long)kk * N_ + r) * N_;
  unsigned short* dst = Ksub + ((long)kk * B_ + i) * B_;
  for (int j = threadIdx.x; j < B_; j += blockDim.x)
    dst[j] = f2bf(src[n_id[j]]);
}

// ---------------------------------------------------------------------------
// 3) grad GEMM:  ypart[seg, i, kk*DIN + d] = sum_{j in seg} Ksub[kk,i,j]*y[j,d]
//    One 16x16 M-tile per wave, all DIN/16 N-tiles accumulated in registers,
//    K chunk of B_/KSPLIT per block (deterministic partials).
//    A fragment: lanes m / m+16 read 4x16B = 64 contiguous bytes per Ksub row.
//    B fragment: yT[d][j] row-major-in-K -> two contiguous 16B loads per lane.
// ---------------------------------------------------------------------------
template <int DIN, int MTOT>
__global__ void k_grad_gemm(const unsigned short* __restrict__ Ksub,
                            const unsigned short* __restrict__ yT,   // [DIN][B_]
                            float* __restrict__ ypart) {             // [KSPLIT][MTOT][KN_*DIN]
  constexpr int NT = DIN / 16;
  int lane  = threadIdx.x & 31;
  int w     = threadIdx.x >> 5;
  int mtile = blockIdx.x * 8 + w;                 // 256 threads = 8 waves
  int kk    = blockIdx.y;
  int seg   = blockIdx.z;
  int i0    = mtile * 16;
  int half  = lane >> 4;
  int l15   = lane & 15;

  const unsigned short* A0 = Ksub + ((long)kk * B_ + (i0 + l15)) * B_;
  int j0 = seg * (B_ / KSPLIT);

  v8f c[NT] = {};
  for (int j = j0; j < j0 + B_ / KSPLIT; j += 32) {
    FragBf a;
    a.h[0] = *(const u16x8*)(A0 + j +      (half ? 8 : 0));   // K 0..7  / 8..15
    a.h[1] = *(const u16x8*)(A0 + j + 16 + (half ? 8 : 0));   // K 16..23 / 24..31
#pragma unroll
    for (int t = 0; t < NT; ++t) {
      const unsigned short* Bp = yT + (long)(t * 16 + l15) * B_ + j + (half ? 8 : 0);
      FragBf b;
      b.h[0] = *(const u16x8*)(Bp);
      b.h[1] = *(const u16x8*)(Bp + 16);
      c[t] = __builtin_amdgcn_wmma_f32_16x16x32_bf16(false, a.v, false, b.v,
                                                     (short)0, c[t], false, false);
    }
  }
  float* yp = ypart + (long)seg * MTOT * (KN_ * DIN) + (long)kk * DIN;
#pragma unroll
  for (int t = 0; t < NT; ++t)
#pragma unroll
    for (int v = 0; v < 8; ++v) {
      int m = v + (half ? 8 : 0);
      yp[(long)(i0 + m) * (KN_ * DIN) + t * 16 + l15] = c[t][v];
    }
}

// reduce layer-1 partials -> bf16 transposed operand + out columns [32,128)
__global__ void k_ep1(const float* __restrict__ y1p, unsigned short* __restrict__ y1T,
                      float* __restrict__ out) {
  int idx = blockIdx.x * blockDim.x + threadIdx.x;
  if (idx >= B_ * D1) return;
  int i = idx / D1, c = idx % D1;
  float v = 0.f;
#pragma unroll
  for (int s = 0; s < KSPLIT; ++s) v += y1p[(long)s * B_ * D1 + idx];
  y1T[c * B_ + i] = f2bf(v);
  if (i < S1_) out[i * DOUT + D_ + c] = v;
}

// reduce layer-2 partials -> out columns [128,416)
__global__ void k_ep2(const float* __restrict__ y2p, float* __restrict__ out) {
  int idx = blockIdx.x * blockDim.x + threadIdx.x;
  if (idx >= S1_ * D2) return;
  int i = idx / D2, c = idx % D2;
  float v = 0.f;
#pragma unroll
  for (int s = 0; s < KSPLIT; ++s) v += y2p[(long)s * S1_ * D2 + idx];
  out[i * DOUT + D_ + D1 + c] = v;
}

// ---------------------------------------------------------------------------
// 4) fp32 MLP via V_WMMA_F32_16X16X4_F32.
//    A 16x4 layout: lane m<16 holds (K=k,k+1), lane m+16 holds (K=k+2,k+3)
//    -> one contiguous float2 per lane.  B 4x16: two strided scalar loads.
// ---------------------------------------------------------------------------
template <int K, int Nn, int NGROUP, bool RELU>
__global__ void k_mlp(const float* __restrict__ A, const float* __restrict__ Bw,
                      const float* __restrict__ bias, float* __restrict__ O) {
  constexpr int NGROUPS = Nn / (16 * NGROUP);
  int lane = threadIdx.x & 31;
  int w    = threadIdx.x >> 5;
  int t    = blockIdx.x * 8 + w;
  int mtile = t / NGROUPS, ng = t % NGROUPS;
  int i0   = mtile * 16;
  int half = lane >> 4;
  int l15  = lane & 15;

  const float* Ar = A + (long)(i0 + l15) * K + (half ? 2 : 0);
  v8f c[NGROUP] = {};
  for (int k = 0; k < K; k += 4) {
    v2f a = *(const v2f*)(Ar + k);
#pragma unroll
    for (int g = 0; g < NGROUP; ++g) {
      int n = (ng * NGROUP + g) * 16 + l15;
      v2f b;
      b.x = Bw[(long)(k +     (half ? 2 : 0)) * Nn + n];
      b.y = Bw[(long)(k + 1 + (half ? 2 : 0)) * Nn + n];
      c[g] = __builtin_amdgcn_wmma_f32_16x16x4_f32(false, a, false, b,
                                                   (short)0, c[g], false, false);
    }
  }
#pragma unroll
  for (int g = 0; g < NGROUP; ++g)
#pragma unroll
    for (int v = 0; v < 8; ++v) {
      int m = v + (half ? 8 : 0);
      int n = (ng * NGROUP + g) * 16 + l15;
      float val = c[g][v] + bias[n];
      if (RELU) val = fmaxf(val, 0.f);
      O[(long)(i0 + m) * Nn + n] = val;
    }
}

// ---------------------------------------------------------------------------
extern "C" void kernel_launch(void* const* d_in, const int* in_sizes, int n_in,
                              void* d_out, int out_size, void* d_ws, size_t ws_size,
                              hipStream_t stream) {
  const float* x       = (const float*)d_in[0];
  const float* L       = (const float*)d_in[1];
  const float* kernels = (const float*)d_in[2];
  const float* W1      = (const float*)d_in[3];
  const float* b1      = (const float*)d_in[4];
  const float* W2      = (const float*)d_in[5];
  const float* b2      = (const float*)d_in[6];
  const int*   n_id    = (const int*)d_in[7];

  float* enc = (float*)d_out;                 // [S1, E]
  float* out = enc + (size_t)S1_ * E_;        // [S1, 416]

  char* ws = (char*)d_ws;
  size_t off = 0;
  auto alloc = [&](size_t bytes) -> void* {
    void* p = ws + off;
    off += (bytes + 255) & ~(size_t)255;
    return p;
  };
  unsigned short* xT   = (unsigned short*)alloc((size_t)D_ * B_ * 2);        // 256 KB
  unsigned short* y1T  = (unsigned short*)alloc((size_t)D1 * B_ * 2);        // 768 KB
  float* y1p           = (float*)alloc((size_t)KSPLIT * B_  * D1 * 4);       // 6 MB
  float* y2p           = (float*)alloc((size_t)KSPLIT * S1_ * D2 * 4);       // 9.4 MB
  float* h             = (float*)alloc((size_t)S1_ * H_ * 4);                // 4 MB
  unsigned short* Ksub = (unsigned short*)alloc((size_t)KN_ * B_ * B_ * 2);  // 100.7 MB

  // 1) diffusion + gather + order-0 norm
  k_diffuse<<<B_ / 8, 256, 0, stream>>>(x, L, n_id, xT, out);
  // 2) bf16 Ksub materialization (L2-resident)
  k_gather<<<dim3(B_, KN_), 256, 0, stream>>>(kernels, n_id, Ksub);
  // 3) grad layer 1: [3,4096,4096] x [4096,32]
  k_grad_gemm<32, B_><<<dim3(B_ / 128, KN_, KSPLIT), 256, 0, stream>>>(Ksub, xT, y1p);
  k_ep1<<<(B_ * D1 + 255) / 256, 256, 0, stream>>>(y1p, y1T, out);
  // 4) grad layer 2: only the S1 rows consumed by `out`
  k_grad_gemm<96, S1_><<<dim3(S1_ / 128, KN_, KSPLIT), 256, 0, stream>>>(Ksub, y1T, y2p);
  k_ep2<<<(S1_ * D2 + 255) / 256, 256, 0, stream>>>(y2p, out);
  // 5) MLP encoder on fp32 WMMA: relu(out@W1+b1)@W2+b2
  k_mlp<DOUT, H_, 4, true ><<<(S1_ / 16) * (H_ / 64) / 8, 256, 0, stream>>>(out, W1, b1, h);
  k_mlp<H_,   E_, 4, false><<<(S1_ / 16) * (E_ / 64) / 8, 256, 0, stream>>>(h, W2, b2, enc);
}
